// SlotAttentionModExt_58076547776895
// MI455X (gfx1250) — matrix operand order repaired and checked by reading, not compiled, not run on hardware
//
#include <hip/hip_runtime.h>

// ---------------------------------------------------------------------------
// Slot Attention for MI455X (gfx1250, wave32, WMMA 16x16x32 f16, f32 accum)
// B=32, N=4096, NS=16, D=512, H=8, DH=64, DFF=2048, ITERS=3
// Software-pipelined fragment loads: double-buffered B frags so memory
// latency overlaps v_wmma issue instead of s_wait_loadcnt 0 before each MACC.
// ---------------------------------------------------------------------------

typedef _Float16 h16;
typedef __attribute__((ext_vector_type(16))) _Float16 v16h;
typedef __attribute__((ext_vector_type(8)))  float     v8f;
typedef __attribute__((ext_vector_type(4)))  float     vf4;

union HFrag { v16h h; vf4 f[2]; };

__device__ __forceinline__ v8f wmma16(v16h a, v16h b, v8f c) {
  // v_wmma_f32_16x16x32_f16
  return __builtin_amdgcn_wmma_f32_16x16x32_f16(false, a, false, b, (short)0, c,
                                                false, false);
}

// A-fragment (16x32, f16) from row-major source with leading dim (halves).
// lane L: row = L&15 ; K chunks [kb, kb+8) and [kb+16, kb+24), kb=(L>>4)*8
__device__ __forceinline__ v16h loadA(const h16* __restrict__ base, int ld,
                                      int rowA, int k0, int kb) {
  HFrag a;
  const h16* p = base + (size_t)rowA * ld + k0 + kb;
  a.f[0] = *(const vf4*)p;
  a.f[1] = *(const vf4*)(p + 16);
  return a.h;
}

// B-fragment (32x16, f16): column n = n0+(L&15) contiguous in K.
// lane L: K chunks [kb, kb+8) and [kb+16, kb+24), kb=k0+(L>>4)*8
__device__ __forceinline__ v16h loadB(const h16* __restrict__ colbase, size_t ldc,
                                      int L, int k0) {
  HFrag b;
  const h16* p = colbase + (size_t)(L & 15) * ldc + k0 + ((L >> 4) * 8);
  b.f[0] = *(const vf4*)p;
  b.f[1] = *(const vf4*)(p + 16);
  return b.h;
}

// ---------------------------------------------------------------------------
// Weight conversion kernels (once per launch)
// ---------------------------------------------------------------------------
__global__ void k_cvt(const float* __restrict__ src, h16* __restrict__ dst,
                      int n, float scale) {
  int i = blockIdx.x * 256 + threadIdx.x;
  if (i < n) dst[i] = (h16)(src[i] * scale);
}

// dst[c*rows + r] = src[r*cols + c]   (transpose + convert)
__global__ void k_cvt_t(const float* __restrict__ src, h16* __restrict__ dst,
                        int rows, int cols) {
  int i = blockIdx.x * 256 + threadIdx.x;
  if (i < rows * cols) {
    int r = i / cols, c = i % cols;
    dst[(size_t)c * rows + r] = (h16)src[i];
  }
}

__global__ void k_zero(float* __restrict__ p, int n) {
  int i = blockIdx.x * 256 + threadIdx.x;
  if (i < n) p[i] = 0.f;
}

// ---------------------------------------------------------------------------
// Kernel 1: LayerNorm(inputs) + K/V projection.  grid = 8192, block = 256.
// k stored [b*4096+n][512] f16 ; v stored transposed vT[b][h][d][4096] f16.
// ---------------------------------------------------------------------------
__global__ __launch_bounds__(256) void k_ln_kv(
    const float* __restrict__ x, const float* __restrict__ g,
    const float* __restrict__ be, const h16* __restrict__ wk,
    const h16* __restrict__ wv, h16* __restrict__ kout, h16* __restrict__ vT) {
  __shared__ __align__(16) h16 sX[16][520];
  const int m0 = blockIdx.x * 16;
  const int bI = m0 >> 12;         // / 4096
  const int jB = m0 & 4095;
  const int tid = threadIdx.x, L = tid & 31, w = tid >> 5;

  // LayerNorm: wave w handles rows 2w, 2w+1 (whole wave per row)
  for (int rr = 0; rr < 2; rr++) {
    int row = w * 2 + rr;
    const float* xr = x + (size_t)(m0 + row) * 512;
    float s = 0.f, s2 = 0.f, vals[16];
#pragma unroll
    for (int e = 0; e < 16; e++) {
      float v0 = xr[L + e * 32]; vals[e] = v0; s += v0; s2 += v0 * v0;
    }
#pragma unroll
    for (int d = 1; d < 32; d <<= 1) {
      s += __shfl_xor(s, d, 32); s2 += __shfl_xor(s2, d, 32);
    }
    float mu = s * (1.f / 512.f);
    float rs = rsqrtf(s2 * (1.f / 512.f) - mu * mu + 1e-5f);
#pragma unroll
    for (int e = 0; e < 16; e++) {
      int c = L + e * 32;
      sX[row][c] = (h16)(((vals[e] - mu) * rs) * g[c] + be[c]);
    }
  }
  __syncthreads();

  const h16* W = (w < 4) ? wk : wv;
  const int wl = w & 3;            // 0..3 within matrix
  const int kb = (L >> 4) * 8, rowA = L & 15;
  v8f acc[8];
#pragma unroll
  for (int t = 0; t < 8; t++) acc[t] = (v8f){};

  // Software-pipelined GEMM: A double-buffered per k-step, B double-buffered
  // per (k-step, tile) so global loads overlap v_wmma issue.
  v16h aF[2], bF[2];
  aF[0] = loadA(&sX[0][0], 520, rowA, 0, kb);
  bF[0] = loadB(W + (size_t)(wl * 8 * 16) * 512, 512, L, 0);
#pragma unroll
  for (int ks = 0; ks < 16; ks++) {
    if (ks + 1 < 16)
      aF[(ks + 1) & 1] = loadA(&sX[0][0], 520, rowA, (ks + 1) * 32, kb);
#pragma unroll
    for (int t = 0; t < 8; t++) {
      int ni = ks * 8 + t + 1;
      if (ni < 128) {
        int nks = ni >> 3, nt = ni & 7;
        bF[ni & 1] =
            loadB(W + (size_t)((wl * 8 + nt) * 16) * 512, 512, L, nks * 32);
      }
      acc[t] = wmma16(aF[ks & 1], bF[(ks * 8 + t) & 1], acc[t]);
    }
  }

  if (w < 4) {                     // K, row-major
#pragma unroll
    for (int t = 0; t < 8; t++) {
      int o = (wl * 8 + t) * 16 + (L & 15);
#pragma unroll
      for (int r = 0; r < 8; r++)
        kout[(size_t)(m0 + r + (L >> 4) * 8) * 512 + o] = (h16)acc[t][r];
    }
  } else {                         // V, transposed: vT[b][h][d][n]
#pragma unroll
    for (int t = 0; t < 8; t++) {
      int o = (wl * 8 + t) * 16 + (L & 15);
      size_t base = (((size_t)bI * 8 + (o >> 6)) * 64 + (o & 63)) * 4096 + jB;
#pragma unroll
      for (int r = 0; r < 8; r++) vT[base + r + (L >> 4) * 8] = (h16)acc[t][r];
    }
  }
}

// vsum[b*512 + h*64+d] = sum_n vT[b][h][d][n].  grid = 2048, block = 256.
__global__ __launch_bounds__(256) void k_vsum(const h16* __restrict__ vT,
                                              float* __restrict__ vsum) {
  int row = blockIdx.x * 8 + (threadIdx.x >> 5);
  int L = threadIdx.x & 31;
  const h16* p = vT + (size_t)row * 4096;
  float s = 0.f;
  for (int e = 0; e < 16; e++) {
    const h16* q = p + (e * 32 + L) * 8;
#pragma unroll
    for (int k = 0; k < 8; k++) s += (float)q[k];
  }
#pragma unroll
  for (int d = 1; d < 32; d <<= 1) s += __shfl_xor(s, d, 32);
  if (L == 0) vsum[row] = s;
}

// LayerNorm rows of [nrows,512] f32 -> f16.  One wave per row.
__global__ __launch_bounds__(256) void k_ln_rows(
    const float* __restrict__ src, const float* __restrict__ g,
    const float* __restrict__ be, h16* __restrict__ dst, int nrows) {
  int row = blockIdx.x * 8 + (threadIdx.x >> 5);
  int L = threadIdx.x & 31;
  if (row >= nrows) return;
  const float* xr = src + (size_t)row * 512;
  float s = 0.f, s2 = 0.f, vals[16];
#pragma unroll
  for (int e = 0; e < 16; e++) {
    float v0 = xr[L + e * 32]; vals[e] = v0; s += v0; s2 += v0 * v0;
  }
#pragma unroll
  for (int d = 1; d < 32; d <<= 1) {
    s += __shfl_xor(s, d, 32); s2 += __shfl_xor(s2, d, 32);
  }
  float mu = s * (1.f / 512.f);
  float rs = rsqrtf(s2 * (1.f / 512.f) - mu * mu + 1e-5f);
  h16* dr = dst + (size_t)row * 512;
#pragma unroll
  for (int e = 0; e < 16; e++) {
    int c = L + e * 32;
    dr[c] = (h16)(((vals[e] - mu) * rs) * g[c] + be[c]);
  }
}

// q = s_ln @ WqT (scale pre-folded).  grid = 32 (b), block = 256.
__global__ __launch_bounds__(256) void k_qproj(const h16* __restrict__ sln,
                                               const h16* __restrict__ wq,
                                               h16* __restrict__ q) {
  int b = blockIdx.x, tid = threadIdx.x, L = tid & 31, w = tid >> 5;
  const h16* A = sln + (size_t)b * 16 * 512;
  const int kb = (L >> 4) * 8, rowA = L & 15;
  v8f acc[4];
#pragma unroll
  for (int t = 0; t < 4; t++) acc[t] = (v8f){};
  for (int ks = 0; ks < 16; ks++) {
    v16h a = loadA(A, 512, rowA, ks * 32, kb);
    v16h bf[4];
#pragma unroll
    for (int t = 0; t < 4; t++)
      bf[t] = loadB(wq + (size_t)((w * 4 + t) * 16) * 512, 512, L, ks * 32);
#pragma unroll
    for (int t = 0; t < 4; t++) acc[t] = wmma16(a, bf[t], acc[t]);
  }
  h16* qb = q + (size_t)b * 16 * 512;
#pragma unroll
  for (int t = 0; t < 4; t++) {
    int o = (w * 4 + t) * 16 + (L & 15);
#pragma unroll
    for (int r = 0; r < 8; r++) qb[(r + (L >> 4) * 8) * 512 + o] = (h16)acc[t][r];
  }
}

// ---------------------------------------------------------------------------
// Fused attention: dots (WMMA) -> cross-head softmax -> attn (f16 LDS)
// -> partial S, attn_mean, updates (WMMA) with global f32 atomics.
// grid = (32 j-tiles, 32 b), block = 256 (wave == head).
// ---------------------------------------------------------------------------
__global__ __launch_bounds__(256) void k_attn(
    const h16* __restrict__ q, const h16* __restrict__ kmat,
    const h16* __restrict__ vT, float* __restrict__ U, float* __restrict__ S,
    float* __restrict__ attn_mean, int writeMean) {
  __shared__ __align__(16) h16 sAttn[128][136];
  __shared__ float sStat[8][128];
  __shared__ float sRed[128];
  const int bI = blockIdx.y, j0 = blockIdx.x * 128;
  const int tid = threadIdx.x, L = tid & 31, hd = tid >> 5;
  const int kb = (L >> 4) * 8, ksB = (L >> 4) * 16;

  // Q fragments for this head (K=64 => 2 ksteps), reused across all j tiles
  const h16* qh = q + ((size_t)bI * 16 + (L & 15)) * 512 + hd * 64;
  HFrag a0, a1;
  a0.f[0] = *(const vf4*)(qh + kb);      a0.f[1] = *(const vf4*)(qh + kb + 16);
  a1.f[0] = *(const vf4*)(qh + 32 + kb); a1.f[1] = *(const vf4*)(qh + 32 + kb + 16);

  // dots: 8 column tiles of j, B frags double-buffered across tiles
  v8f acc[8];
  const h16* kbase = kmat + ((size_t)bI * 4096 + j0) * 512 + hd * 64;
  v16h kb0[2], kb1[2];
  kb0[0] = loadB(kbase, 512, L, 0);
  kb1[0] = loadB(kbase, 512, L, 32);
#pragma unroll
  for (int t = 0; t < 8; t++) {
    if (t + 1 < 8) {
      const h16* nb = kbase + (size_t)((t + 1) * 16) * 512;
      kb0[(t + 1) & 1] = loadB(nb, 512, L, 0);
      kb1[(t + 1) & 1] = loadB(nb, 512, L, 32);
    }
    v8f c = (v8f){};
    c = wmma16(a0.h, kb0[t & 1], c);
    c = wmma16(a1.h, kb1[t & 1], c);
    acc[t] = c;
  }

  // start pulling V tiles toward the caches while we do softmax
  const h16* vbase = vT + (((size_t)bI * 8 + hd) * 64) * 4096 + j0;
#pragma unroll
  for (int t = 0; t < 4; t++)
    __builtin_prefetch(vbase + (size_t)(t * 16 + (L & 15)) * 4096 + ksB, 0, 1);

  // per-head column max (rows split lane L <-> L+16)
#pragma unroll
  for (int t = 0; t < 8; t++) {
    float m = acc[t][0];
#pragma unroll
    for (int r = 1; r < 8; r++) m = fmaxf(m, acc[t][r]);
    m = fmaxf(m, __shfl_xor(m, 16, 32));
    if (L < 16) sStat[hd][t * 16 + L] = m;
  }
  __syncthreads();
  if (tid < 128) {
    float m = sStat[0][tid];
#pragma unroll
    for (int hh = 1; hh < 8; hh++) m = fmaxf(m, sStat[hh][tid]);
    sRed[tid] = m;
  }
  __syncthreads();
  // exp + per-head column sums
#pragma unroll
  for (int t = 0; t < 8; t++) {
    float m = sRed[t * 16 + (L & 15)];
#pragma unroll
    for (int r = 0; r < 8; r++) acc[t][r] = __expf(acc[t][r] - m);
  }
#pragma unroll
  for (int t = 0; t < 8; t++) {
    float s = 0.f;
#pragma unroll
    for (int r = 0; r < 8; r++) s += acc[t][r];
    s += __shfl_xor(s, 16, 32);
    if (L < 16) sStat[hd][t * 16 + L] = s;   // safe: old reads done pre-barrier
  }
  __syncthreads();
  if (tid < 128) {
    float s = 0.f;
#pragma unroll
    for (int hh = 0; hh < 8; hh++) s += sStat[hh][tid];
    sRed[tid] = 1.f / s;
  }
  __syncthreads();
  // normalize; store attn (pre-renorm softmax) to LDS as f16
#pragma unroll
  for (int t = 0; t < 8; t++) {
    float inv = sRed[t * 16 + (L & 15)];
#pragma unroll
    for (int r = 0; r < 8; r++)
      sAttn[hd * 16 + r + (L >> 4) * 8][t * 16 + (L & 15)] =
          (h16)(acc[t][r] * inv);
  }
  __syncthreads();

  // partial S[b][h*16+i] += sum_j attn
  if (tid < 128) {
    const h16* rp = sAttn[tid];
    float s = 0.f;
    for (int j = 0; j < 128; j++) s += (float)rp[j];
    atomicAdd(&S[bI * 128 + tid], s);
  }
  // attn_mean (last iteration only): mean over heads of pre-renorm attn
  if (writeMean) {
#pragma unroll
    for (int rep = 0; rep < 8; rep++) {
      int idx = tid + rep * 256, i = idx >> 7, j = idx & 127;
      float s = 0.f;
#pragma unroll
      for (int hh = 0; hh < 8; hh++) s += (float)sAttn[hh * 16 + i][j];
      attn_mean[((size_t)bI * 16 + i) * 4096 + j0 + j] = s * 0.125f;
    }
  }

  // updates: U[i][h][d] += attn_h[16 x 128] @ V_h[128 x 64]
  // A (LDS) per k-step + B (global) double-buffered pipeline.
  v8f uacc[4];
#pragma unroll
  for (int t = 0; t < 4; t++) uacc[t] = (v8f){};
  v16h ua[2], ub[2];
  ua[0] = loadA(&sAttn[hd * 16][0], 136, L & 15, 0, kb);
  ub[0] = loadB(vbase, 4096, L, 0);
#pragma unroll
  for (int ks = 0; ks < 4; ks++) {
    if (ks + 1 < 4)
      ua[(ks + 1) & 1] = loadA(&sAttn[hd * 16][0], 136, L & 15, (ks + 1) * 32, kb);
#pragma unroll
    for (int t = 0; t < 4; t++) {
      int ni = ks * 4 + t + 1;
      if (ni < 16) {
        int nks = ni >> 2, nt = ni & 3;
        ub[ni & 1] = loadB(vbase + (size_t)(nt * 16) * 4096, 4096, L, nks * 32);
      }
      uacc[t] = wmma16(ua[ks & 1], ub[(ks * 4 + t) & 1], uacc[t]);
    }
  }
#pragma unroll
  for (int t = 0; t < 4; t++) {
    int d = t * 16 + (L & 15);
#pragma unroll
    for (int r = 0; r < 8; r++) {
      int i = r + (L >> 4) * 8;
      atomicAdd(&U[(((size_t)bI * 16 + i) * 8 + hd) * 64 + d], uacc[t][r]);
    }
  }
}

// ---------------------------------------------------------------------------
// Post kernel per batch: renorm -> Wo + residual -> LN -> FF1(relu) -> FF2
// + residual.  grid = 32, block = 256, dynamic LDS (131840 B).
// ---------------------------------------------------------------------------
__global__ __launch_bounds__(256) void k_post(
    const float* __restrict__ U, const float* __restrict__ S,
    const float* __restrict__ vsum, const float* __restrict__ slots_in,
    const float* __restrict__ lnf_g, const float* __restrict__ lnf_b,
    const h16* __restrict__ wo, const h16* __restrict__ w1t,
    const float* __restrict__ b1, const h16* __restrict__ w2t,
    const float* __restrict__ b2, float* __restrict__ slots_out,
    float* __restrict__ out_final, int writeOut) {
  extern __shared__ __align__(16) char dsm[];
  h16*   sUp  = (h16*)dsm;                                 // [16][520]
  float* sMid = (float*)(dsm + 16 * 520 * 2);              // [16][512]
  h16*   sLn  = (h16*)(dsm + 16 * 520 * 2 + 16 * 512 * 4); // [16][520]
  h16*   sHid = (h16*)(dsm + 16 * 520 * 4 + 16 * 512 * 4); // [16][2056]
  const int b = blockIdx.x, tid = threadIdx.x, L = tid & 31, w = tid >> 5;
  const int kb = (L >> 4) * 8, rowA = L & 15;

  // 1. renormed updates -> f16
#pragma unroll
  for (int rep = 0; rep < 32; rep++) {
    int idx = tid + rep * 256, i = idx >> 9, c = idx & 511;
    float denom = S[b * 128 + (c >> 6) * 16 + i] + 4096.f * 1e-8f;
    float val = (U[((size_t)b * 16 + i) * 512 + c] +
                 1e-8f * vsum[(size_t)b * 512 + c]) / denom;
    sUp[i * 520 + c] = (h16)val;
  }
  __syncthreads();

  // 2. sMid = sUp @ WoT + slots_prev  (batch-load 4 B frags per k-step)
  {
    v8f acc[4];
#pragma unroll
    for (int t = 0; t < 4; t++) acc[t] = (v8f){};
    for (int ks = 0; ks < 16; ks++) {
      v16h a = loadA(sUp, 520, rowA, ks * 32, kb);
      v16h bf[4];
#pragma unroll
      for (int t = 0; t < 4; t++)
        bf[t] = loadB(wo + (size_t)((w * 4 + t) * 16) * 512, 512, L, ks * 32);
#pragma unroll
      for (int t = 0; t < 4; t++) acc[t] = wmma16(a, bf[t], acc[t]);
    }
#pragma unroll
    for (int t = 0; t < 4; t++) {
      int o = (w * 4 + t) * 16 + (L & 15);
#pragma unroll
      for (int r = 0; r < 8; r++) {
        int i = r + (L >> 4) * 8;
        sMid[i * 512 + o] = acc[t][r] + slots_in[((size_t)b * 16 + i) * 512 + o];
      }
    }
  }
  __syncthreads();

  // 3. LayerNorm(sMid) -> sLn f16 ; one wave per 2 rows
  for (int rr = 0; rr < 2; rr++) {
    int row = w * 2 + rr;
    float s = 0.f, s2 = 0.f, vals[16];
#pragma unroll
    for (int e = 0; e < 16; e++) {
      float v0 = sMid[row * 512 + L + e * 32]; vals[e] = v0; s += v0; s2 += v0 * v0;
    }
#pragma unroll
    for (int d = 1; d < 32; d <<= 1) {
      s += __shfl_xor(s, d, 32); s2 += __shfl_xor(s2, d, 32);
    }
    float mu = s * (1.f / 512.f);
    float rs = rsqrtf(s2 * (1.f / 512.f) - mu * mu + 1e-5f);
#pragma unroll
    for (int e = 0; e < 16; e++) {
      int c = L + e * 32;
      sLn[row * 520 + c] = (h16)(((vals[e] - mu) * rs) * lnf_g[c] + lnf_b[c]);
    }
  }
  __syncthreads();

  // 4. sHid = relu(sLn @ W1 + b1) ; W1T[f][d], 16 f-tiles per wave,
  //    B frags double-buffered over the k-loop.
  for (int tt = 0; tt < 16; tt++) {
    int f0 = (w * 16 + tt) * 16;
    const h16* colbase = w1t + (size_t)f0 * 512;
    v8f c = (v8f){};
    v16h bf[2];
    bf[0] = loadB(colbase, 512, L, 0);
#pragma unroll
    for (int ks = 0; ks < 16; ks++) {
      if (ks + 1 < 16) bf[(ks + 1) & 1] = loadB(colbase, 512, L, (ks + 1) * 32);
      v16h a = loadA(sLn, 520, rowA, ks * 32, kb);
      c = wmma16(a, bf[ks & 1], c);
    }
    float bb = b1[f0 + (L & 15)];
#pragma unroll
    for (int r = 0; r < 8; r++) {
      float vv = c[r] + bb;
      sHid[(r + (L >> 4) * 8) * 2056 + f0 + (L & 15)] = (h16)(vv > 0.f ? vv : 0.f);
    }
  }
  __syncthreads();

  // 5. out = sHid @ W2 + b2 + sMid ; W2T[o][f] (batch 4 B frags per k-step)
  {
    v8f acc[4];
#pragma unroll
    for (int t = 0; t < 4; t++) acc[t] = (v8f){};
    for (int ks = 0; ks < 64; ks++) {
      v16h a = loadA(sHid, 2056, rowA, ks * 32, kb);
      v16h bf[4];
#pragma unroll
      for (int t = 0; t < 4; t++)
        bf[t] = loadB(w2t + (size_t)((w * 4 + t) * 16) * 2048, 2048, L, ks * 32);
#pragma unroll
      for (int t = 0; t < 4; t++) acc[t] = wmma16(a, bf[t], acc[t]);
    }
#pragma unroll
    for (int t = 0; t < 4; t++) {
      int o = (w * 4 + t) * 16 + (L & 15);
#pragma unroll
      for (int r = 0; r < 8; r++) {
        int i = r + (L >> 4) * 8;
        float vv = acc[t][r] + b2[o] + sMid[i * 512 + o];
        slots_out[((size_t)b * 16 + i) * 512 + o] = vv;
        if (writeOut) out_final[((size_t)b * 16 + i) * 512 + o] = vv;
      }
    }
  }
}

// ---------------------------------------------------------------------------
extern "C" void kernel_launch(void* const* d_in, const int* in_sizes, int n_in,
                              void* d_out, int out_size, void* d_ws,
                              size_t ws_size, hipStream_t stream) {
  (void)in_sizes; (void)n_in; (void)out_size; (void)ws_size;
  const float* x      = (const float*)d_in[0];
  const float* cond   = (const float*)d_in[1];
  const float* ln_in_g = (const float*)d_in[2];
  const float* ln_in_b = (const float*)d_in[3];
  const float* Wk = (const float*)d_in[4];
  const float* Wv = (const float*)d_in[5];
  const float* Wq = (const float*)d_in[6];
  const float* Wo = (const float*)d_in[7];
  const float* ln_s_g = (const float*)d_in[8];
  const float* ln_s_b = (const float*)d_in[9];
  const float* ln_f_g = (const float*)d_in[10];
  const float* ln_f_b = (const float*)d_in[11];
  const float* W1 = (const float*)d_in[12];
  const float* b1 = (const float*)d_in[13];
  const float* W2 = (const float*)d_in[14];
  const float* b2 = (const float*)d_in[15];

  float* out_slots = (float*)d_out;                       // [32,16,512]
  float* out_amean = (float*)d_out + 32 * 16 * 512;       // [32,16,4096]

  size_t off = 0;
  char* wsb = (char*)d_ws;
  auto alloc = [&](size_t bytes) {
    void* p = wsb + off; off += (bytes + 255) & ~(size_t)255; return p;
  };
  h16* kbuf  = (h16*)alloc((size_t)32 * 4096 * 512 * 2);
  h16* vT    = (h16*)alloc((size_t)32 * 4096 * 512 * 2);
  h16* wk16  = (h16*)alloc(512 * 512 * 2);
  h16* wv16  = (h16*)alloc(512 * 512 * 2);
  h16* wq16  = (h16*)alloc(512 * 512 * 2);
  h16* wo16  = (h16*)alloc(512 * 512 * 2);
  h16* w1t   = (h16*)alloc((size_t)512 * 2048 * 2);
  h16* w2t   = (h16*)alloc((size_t)2048 * 512 * 2);
  h16* qbuf  = (h16*)alloc(32 * 16 * 512 * 2);
  h16* slnb  = (h16*)alloc(32 * 16 * 512 * 2);
  float* slots = (float*)alloc(32 * 16 * 512 * 4);
  float* U     = (float*)alloc(32 * 16 * 512 * 4);   // 1 MiB, 256-aligned
  float* S     = (float*)alloc(32 * 128 * 4);        // directly after U
  float* vsum  = (float*)alloc(32 * 512 * 4);

  // --- prep ---
  k_cvt<<<1024, 256, 0, stream>>>(Wk, wk16, 512 * 512, 1.f);
  k_cvt<<<1024, 256, 0, stream>>>(Wv, wv16, 512 * 512, 1.f);
  k_cvt<<<1024, 256, 0, stream>>>(Wq, wq16, 512 * 512, 0.125f); // fold DH^-0.5
  k_cvt<<<1024, 256, 0, stream>>>(Wo, wo16, 512 * 512, 1.f);
  k_cvt_t<<<4096, 256, 0, stream>>>(W1, w1t, 512, 2048);  // w1t[f][d]
  k_cvt_t<<<4096, 256, 0, stream>>>(W2, w2t, 2048, 512);  // w2t[o][f]
  k_ln_kv<<<8192, 256, 0, stream>>>(x, ln_in_g, ln_in_b, wk16, wv16, kbuf, vT);
  k_vsum<<<2048, 256, 0, stream>>>(vT, vsum);

  const size_t POST_LDS = (size_t)16 * 520 * 2 + 16 * 512 * 4 + 16 * 520 * 2 +
                          (size_t)16 * 2056 * 2;  // 131840 B
  const float* slots_src = cond;
  for (int it = 0; it < 3; it++) {
    int last = (it == 2);
    k_ln_rows<<<64, 256, 0, stream>>>(slots_src, ln_s_g, ln_s_b, slnb, 512);
    k_qproj<<<32, 256, 0, stream>>>(slnb, wq16, qbuf);
    k_zero<<<(32 * 16 * 512 + 32 * 128 + 255) / 256, 256, 0, stream>>>(
        U, 32 * 16 * 512 + 32 * 128);  // U and S are contiguous
    k_attn<<<dim3(32, 32), 256, 0, stream>>>(qbuf, kbuf, vT, U, S, out_amean,
                                             last);
    k_post<<<32, 256, POST_LDS, stream>>>(U, S, vsum, slots_src, ln_f_g,
                                          ln_f_b, wo16, w1t, b1, w2t, b2,
                                          slots, out_slots, last);
    slots_src = slots;
  }
}
